// Quantization_Block_50508815401498
// MI455X (gfx1250) — matrix operand order repaired and checked by reading, not compiled, not run on hardware
//
#include <hip/hip_runtime.h>
#include <stdint.h>

// ---------------------------------------------------------------------------
// VQ nearest-code (argmax quirk preserved) for MI455X / gfx1250.
// bf16x3 split-precision WMMA GEMM (hi*hi + hi*lo + lo*hi) fused with the
// per-row running argmax; B-chunk staging is register-prefetch pipelined so
// L2 latency hides under the WMMA stream. Separate e_sq and gather kernels.
// ---------------------------------------------------------------------------

typedef __bf16 bf16x16 __attribute__((ext_vector_type(16)));
typedef float  f32x8   __attribute__((ext_vector_type(8)));
typedef unsigned short u16x8 __attribute__((ext_vector_type(8)));

#define N_TOKENS 32768
#define K_CODES  8192
#define DIM      512

#define WG_ROWS  128   // rows of x per workgroup
#define WG_CODES 128   // codebook tile width
#define DCHUNK   64    // D staged per B-chunk
#define A_STRIDE 520   // 512 + 8 pad (elements) -> conflict-free b128 LDS reads
#define B_STRIDE 72    // 64 + 8 pad
#define NCHUNKS  ((K_CODES / WG_CODES) * (DIM / DCHUNK))   // 64 * 8 = 512

union FragU {
  u16x8   h[2];
  bf16x16 v;
};

__device__ inline unsigned short f2bf_rne(float f) {
  unsigned u = __float_as_uint(f);
  u += 0x7FFFu + ((u >> 16) & 1u);
  return (unsigned short)(u >> 16);
}
__device__ inline void bf16_split(float f, unsigned short& hi, unsigned short& lo) {
  hi = f2bf_rne(f);
  float fh = __uint_as_float(((unsigned)hi) << 16);
  lo = f2bf_rne(f - fh);
}

// ---------------------------------------------------------------------------
// Kernel 1: e_sq[k] = sum_d codebook[k][d]^2   (one wave per code row)
// ---------------------------------------------------------------------------
__global__ __launch_bounds__(256) void esq_kernel(const float* __restrict__ cb,
                                                  float* __restrict__ esq) {
  const int t = threadIdx.x;
  const int wid = t >> 5, lane = t & 31;
  const int row = blockIdx.x * 8 + wid;
  const float4* p = reinterpret_cast<const float4*>(cb + (size_t)row * DIM);
  float s = 0.f;
#pragma unroll
  for (int i = 0; i < 4; ++i) {
    float4 v = p[lane + 32 * i];
    s += v.x * v.x + v.y * v.y + v.z * v.z + v.w * v.w;
  }
#pragma unroll
  for (int off = 16; off > 0; off >>= 1) s += __shfl_xor(s, off, 32);
  if (lane == 0) esq[row] = s;
}

// ---------------------------------------------------------------------------
// Kernel 2: fused GEMM (bf16x3 WMMA) + per-row argmax of (-2*prod + e_sq)
// Grid: N_TOKENS/WG_ROWS blocks of 256 threads (8 waves, 4 (M) x 2 (N)).
// Wave tile: 32 rows x 64 codes = 2x4 accumulator tiles of 16x16.
// ---------------------------------------------------------------------------
__global__ __launch_bounds__(256, 1) void vq_argmax_kernel(
    const float* __restrict__ x, const float* __restrict__ cb,
    const float* __restrict__ esq, int* __restrict__ best) {
  __shared__ unsigned short Ahi[WG_ROWS * A_STRIDE];
  __shared__ unsigned short Alo[WG_ROWS * A_STRIDE];
  __shared__ unsigned short Bhi[WG_CODES * B_STRIDE];
  __shared__ unsigned short Blo[WG_CODES * B_STRIDE];
  __shared__ float red_v[8][32];
  __shared__ int   red_i[8][32];

  const int t     = threadIdx.x;
  const int wid   = t >> 5;
  const int lane  = t & 31;
  const int l     = lane & 15;   // column / code lane
  const int h     = lane >> 4;   // half-wave
  const int waveM = wid >> 1;    // 0..3
  const int waveN = wid & 1;     // 0..1
  const int rowBase = blockIdx.x * WG_ROWS;

  // Per-thread B staging geometry (8 float4 per thread per chunk).
  // Chunk g: kt = g>>3 (code tile), dc = g&7 (64-dim slice).
  const int bc0 = t >> 4;        // code row handled in step i: bc0 + i*16
  const int bq  = t & 15;        // float4 column within the 64-dim slice

  // ---- Stage A: 128 x 512 fp32 -> hi/lo bf16 in LDS (once) ----
#pragma unroll 4
  for (int i = 0; i < 64; ++i) {
    int j   = i * 256 + t;   // float4 index, 16384 total
    int row = j >> 7;        // 128 float4 per row
    int dq  = j & 127;
    float4 v = reinterpret_cast<const float4*>(x)[(rowBase + row) * (DIM / 4) + dq];
    unsigned short a0, a1, a2, a3, b0, b1, b2, b3;
    bf16_split(v.x, a0, b0); bf16_split(v.y, a1, b1);
    bf16_split(v.z, a2, b2); bf16_split(v.w, a3, b3);
    *reinterpret_cast<ushort4*>(&Ahi[row * A_STRIDE + dq * 4]) = make_ushort4(a0, a1, a2, a3);
    *reinterpret_cast<ushort4*>(&Alo[row * A_STRIDE + dq * 4]) = make_ushort4(b0, b1, b2, b3);
  }

  float bv[2][8];
  int   bi[2][8];
#pragma unroll
  for (int m = 0; m < 2; ++m)
#pragma unroll
    for (int r = 0; r < 8; ++r) { bv[m][r] = -__builtin_inff(); bi[m][r] = 0; }

  // ---- Prefetch chunk 0 into registers ----
  float4 pre[8];
#pragma unroll
  for (int i = 0; i < 8; ++i)
    pre[i] = reinterpret_cast<const float4*>(cb)[(bc0 + i * 16) * (DIM / 4) + bq];

  f32x8 acc[2][4] = {};

  for (int g = 0; g < NCHUNKS; ++g) {
    const int dc = g & 7;

    __syncthreads();  // previous chunk's fragment loads complete
    // ---- Convert prefetched chunk -> hi/lo bf16 in LDS ----
#pragma unroll
    for (int i = 0; i < 8; ++i) {
      float4 v = pre[i];
      int c = bc0 + i * 16;
      unsigned short a0, a1, a2, a3, b0, b1, b2, b3;
      bf16_split(v.x, a0, b0); bf16_split(v.y, a1, b1);
      bf16_split(v.z, a2, b2); bf16_split(v.w, a3, b3);
      *reinterpret_cast<ushort4*>(&Bhi[c * B_STRIDE + bq * 4]) = make_ushort4(a0, a1, a2, a3);
      *reinterpret_cast<ushort4*>(&Blo[c * B_STRIDE + bq * 4]) = make_ushort4(b0, b1, b2, b3);
    }
    // ---- Issue global loads for chunk g+1 (complete during compute) ----
    if (g + 1 < NCHUNKS) {
      const int kt2 = (g + 1) >> 3;
      const int dc2 = (g + 1) & 7;
#pragma unroll
      for (int i = 0; i < 8; ++i)
        pre[i] = reinterpret_cast<const float4*>(
            cb)[(kt2 * WG_CODES + bc0 + i * 16) * (DIM / 4) + dc2 * (DCHUNK / 4) + bq];
    }
    __syncthreads();  // staged chunk visible

    // ---- Compute: 2 d-steps of 32, bf16x3 ----
#pragma unroll
    for (int dstep = 0; dstep < 2; ++dstep) {
      const int d0  = dc * DCHUNK + dstep * 32;  // into A (full-D LDS)
      const int dd0 = dstep * 32;                // into B chunk
      // A fragments: lane(l,h) holds row l, d = d0 + h*8 + {0..7}, d0+16+h*8+{0..7}
      FragU ah[2], al[2];
#pragma unroll
      for (int m = 0; m < 2; ++m) {
        int row  = waveM * 32 + m * 16 + l;
        int base = row * A_STRIDE + d0 + h * 8;
        ah[m].h[0] = *reinterpret_cast<const u16x8*>(&Ahi[base]);
        ah[m].h[1] = *reinterpret_cast<const u16x8*>(&Ahi[base + 16]);
        al[m].h[0] = *reinterpret_cast<const u16x8*>(&Alo[base]);
        al[m].h[1] = *reinterpret_cast<const u16x8*>(&Alo[base + 16]);
      }
      // B fragments: lane(l,h) holds code column l, K = h*8 + {0..7}, 16+h*8+{0..7}
      FragU bh[4], bl[4];
#pragma unroll
      for (int n = 0; n < 4; ++n) {
        int code = waveN * 64 + n * 16 + l;
        int base = code * B_STRIDE + dd0 + h * 8;
        bh[n].h[0] = *reinterpret_cast<const u16x8*>(&Bhi[base]);
        bh[n].h[1] = *reinterpret_cast<const u16x8*>(&Bhi[base + 16]);
        bl[n].h[0] = *reinterpret_cast<const u16x8*>(&Blo[base]);
        bl[n].h[1] = *reinterpret_cast<const u16x8*>(&Blo[base + 16]);
      }
#pragma unroll
      for (int m = 0; m < 2; ++m)
#pragma unroll
        for (int n = 0; n < 4; ++n) {
          acc[m][n] = __builtin_amdgcn_wmma_f32_16x16x32_bf16(
              false, ah[m].v, false, bh[n].v, (short)0, acc[m][n], false, false);
          acc[m][n] = __builtin_amdgcn_wmma_f32_16x16x32_bf16(
              false, ah[m].v, false, bl[n].v, (short)0, acc[m][n], false, false);
          acc[m][n] = __builtin_amdgcn_wmma_f32_16x16x32_bf16(
              false, al[m].v, false, bh[n].v, (short)0, acc[m][n], false, false);
        }
    }

    // ---- End of a code tile: fold into running argmax, reset accumulators ----
    // (codes ascend in kt and n; strict '>' preserves first-max tie-break)
    if (dc == 7) {
      const int kt = g >> 3;
#pragma unroll
      for (int n = 0; n < 4; ++n) {
        int code = kt * WG_CODES + waveN * 64 + n * 16 + l;
        float es = esq[code];
#pragma unroll
        for (int m = 0; m < 2; ++m)
#pragma unroll
          for (int r = 0; r < 8; ++r) {
            float val = fmaf(-2.0f, acc[m][n][r], es);  // x_sq is row-constant
            if (val > bv[m][r]) { bv[m][r] = val; bi[m][r] = code; }
          }
      }
#pragma unroll
      for (int m = 0; m < 2; ++m)
#pragma unroll
        for (int n = 0; n < 4; ++n) acc[m][n] = f32x8{};
    }
  }

  // ---- Cross-lane argmax within each 16-lane half (lower idx wins ties) ----
#pragma unroll
  for (int m = 0; m < 2; ++m)
#pragma unroll
    for (int r = 0; r < 8; ++r) {
      float v = bv[m][r];
      int idx = bi[m][r];
#pragma unroll
      for (int off = 1; off < 16; off <<= 1) {
        float ov = __shfl_xor(v, off, 32);
        int   oi = __shfl_xor(idx, off, 32);
        if (ov > v || (ov == v && oi < idx)) { v = ov; idx = oi; }
      }
      bv[m][r] = v; bi[m][r] = idx;
    }

  __syncthreads();
  if (l == 0) {  // lanes 0 (h=0) and 16 (h=1)
#pragma unroll
    for (int m = 0; m < 2; ++m)
#pragma unroll
      for (int r = 0; r < 8; ++r) {
        int rw = m * 16 + h * 8 + r;
        red_v[wid][rw] = bv[m][r];
        red_i[wid][rw] = bi[m][r];
      }
  }
  __syncthreads();
  // combine the two waveN halves sharing each row, write indices
  if ((wid & 1) == 0) {
    float v0 = red_v[wid][lane];     int i0 = red_i[wid][lane];
    float v1 = red_v[wid + 1][lane]; int i1 = red_i[wid + 1][lane];
    int pick = (v1 > v0 || (v1 == v0 && i1 < i0)) ? i1 : i0;
    best[rowBase + waveM * 32 + lane] = pick;
  }
}

// ---------------------------------------------------------------------------
// Kernel 3: out[n][:] = codebook[best[n]][:]
// ---------------------------------------------------------------------------
__global__ __launch_bounds__(128) void gather_kernel(const float* __restrict__ cb,
                                                     const int* __restrict__ best,
                                                     float* __restrict__ out) {
  const int rowN = blockIdx.x;
  const int idx  = best[rowN];
  const float4* src = reinterpret_cast<const float4*>(cb + (size_t)idx * DIM);
  float4*       dst = reinterpret_cast<float4*>(out + (size_t)rowN * DIM);
  dst[threadIdx.x] = src[threadIdx.x];
}

// ---------------------------------------------------------------------------
extern "C" void kernel_launch(void* const* d_in, const int* in_sizes, int n_in,
                              void* d_out, int out_size, void* d_ws, size_t ws_size,
                              hipStream_t stream) {
  (void)in_sizes; (void)n_in; (void)out_size; (void)ws_size;
  const float* x  = (const float*)d_in[0];
  const float* cb = (const float*)d_in[1];
  float* out = (float*)d_out;
  int*   best = (int*)d_ws;                                     // 32768 ints
  float* esq  = (float*)((char*)d_ws + N_TOKENS * sizeof(int)); // 8192 floats

  esq_kernel<<<K_CODES / 8, 256, 0, stream>>>(cb, esq);
  vq_argmax_kernel<<<N_TOKENS / WG_ROWS, 256, 0, stream>>>(x, cb, esq, best);
  gather_kernel<<<N_TOKENS, 128, 0, stream>>>(cb, best, out);
}
